// ECEmodel_21440476742302
// MI455X (gfx1250) — matrix operand order repaired and checked by reading, not compiled, not run on hardware
//
#include <hip/hip_runtime.h>
#include <hip/hip_bf16.h>

// Problem constants (match reference)
#define B_   64
#define S_   512
#define H_   768
#define L_   32
#define D_   1536            // 2*H
#define ROWS 16              // clause rows per block
#define KSTEP 32             // bf16 WMMA K depth
#define NWAVES 8
#define NTILES 12            // 1536 / (8 waves * 16 cols)
#define ASTRIDE 1544         // 1536 + 8 bf16 pad (bank-conflict avoidance)

typedef __attribute__((ext_vector_type(16))) __bf16 v16bf;
typedef __attribute__((ext_vector_type(8)))  __bf16 v8bf;
typedef __attribute__((ext_vector_type(8)))  float  v8f;

__device__ __forceinline__ __bf16 to_bf16(float f) {
  unsigned u = __builtin_bit_cast(unsigned, f);
  u = u + 0x7FFFu + ((u >> 16) & 1u);           // round-to-nearest-even
  unsigned short h = (unsigned short)(u >> 16);
  return __builtin_bit_cast(__bf16, h);
}

// ---- Pre-pass: W_pool f32 [D,D] -> bf16 in workspace (one-time, 4.5 MB) ----
__global__ __launch_bounds__(256)
void convert_w_kernel(const float* __restrict__ W, unsigned* __restrict__ Wbf2) {
  const int i = (blockIdx.x * 256 + threadIdx.x) * 4;   // 4 f32 per thread
  float4 w = *(const float4*)(W + i);
  unsigned short h0 = __builtin_bit_cast(unsigned short, to_bf16(w.x));
  unsigned short h1 = __builtin_bit_cast(unsigned short, to_bf16(w.y));
  unsigned short h2 = __builtin_bit_cast(unsigned short, to_bf16(w.z));
  unsigned short h3 = __builtin_bit_cast(unsigned short, to_bf16(w.w));
  uint2 o;
  o.x = (unsigned)h0 | ((unsigned)h1 << 16);
  o.y = (unsigned)h2 | ((unsigned)h3 << 16);
  *(uint2*)(Wbf2 + (i >> 1)) = o;               // 8-byte packed store
}

// ---- Fused kernel: segment means -> LDS bf16 A tile, WMMA GEMM, tanh/dot ----
template <bool PRECONV>
__global__ __launch_bounds__(256)
void ece_fused_kernel(const float*  __restrict__ seq,     // [B,S,H]
                      const float*  __restrict__ W_pool,  // [D,D] f32 ([N,K])
                      const __bf16* __restrict__ Wbf,     // [D,D] bf16 (if PRECONV)
                      const float*  __restrict__ b_pool,  // [D]
                      const float*  __restrict__ W_out,   // [D]
                      const float*  __restrict__ b_out1,  // [1]
                      const int*    __restrict__ pos,     // [B,L+1]
                      float* __restrict__ out)            // [B*L]
{
  __shared__ __bf16 As[ROWS][ASTRIDE];          // A tile: 16 x 1536 bf16
  __shared__ float  part[ROWS][NWAVES];

  const int blk  = blockIdx.x;                  // 0..127
  const int b    = blk >> 1;                    // 2 blocks per doc (16 clauses each)
  const int l0   = (blk & 1) * ROWS;
  const int tid  = threadIdx.x;
  const int lane = tid & 31;
  const int wave = tid >> 5;

  // ---------------- Phase 0: segment means + CLS -> LDS (bf16) -------------
  const int* p = pos + b * (L_ + 1);
  const float* srow0 = seq + (size_t)b * S_ * H_;
  for (int r = 0; r < ROWS; ++r) {
    const int l     = l0 + r;
    const int start = p[l] + 1;
    const int end   = p[l + 1] + 1 + (l == L_ - 1 ? 1 : 0); // last clause keeps SEP
    const float inv = 1.0f / (float)(end - start);
    #pragma unroll
    for (int q = 0; q < 3; ++q) {               // 768 = 3 * 256
      const int h = tid + q * 256;
      float s = 0.f;
      for (int t = start; t < end; ++t)
        s += srow0[(size_t)t * H_ + h];
      As[r][h]      = to_bf16(s * inv);         // clause mean
      As[r][H_ + h] = to_bf16(srow0[h]);        // CLS token (t = 0)
    }
  }
  __syncthreads();

  // ---------------- Phase 1: WMMA GEMM (K outer, 12 N-tiles in registers) --
  // A frag (16-bit A 16x32): lanes 0-15 -> K 0..15, lanes 16-31 -> K 16..31
  const int mhalf = lane >> 4;                  // 0 | 1
  const int nlane = lane & 15;
  const int koffA = mhalf * 16;
  const int koffB = mhalf * 16;                 // B frag: 16 contiguous K per lane
  const int am    = nlane;                      // A row held by this lane
  const int nbase = wave * (NTILES * 16);       // this wave's first output column

  // Base pointers for this lane's B column (j advances by a constant offset)
  const __bf16* wb = PRECONV ? (Wbf + (size_t)(nbase + nlane) * D_ + koffB) : nullptr;
  const float*  wf = PRECONV ? nullptr : (W_pool + (size_t)(nbase + nlane) * D_ + koffB);

  // B fragment loader (j-tile, k offset)
  auto loadB = [&](int j, int k) -> v16bf {
    if constexpr (PRECONV) {
      const __bf16* wp = wb + (size_t)j * 16 * D_ + k;    // const 49152B offsets
      v8bf blo = *(const v8bf*)(wp);
      v8bf bhi = *(const v8bf*)(wp + 8);
      return __builtin_shufflevector(blo, bhi,
               0,1,2,3,4,5,6,7,8,9,10,11,12,13,14,15);
    } else {
      const float4* wp4 = (const float4*)(wf + (size_t)j * 16 * D_ + k);
      v16bf bb;
      #pragma unroll
      for (int i = 0; i < 4; ++i) {
        float4 w = wp4[i];
        bb[4*i+0] = to_bf16(w.x); bb[4*i+1] = to_bf16(w.y);
        bb[4*i+2] = to_bf16(w.z); bb[4*i+3] = to_bf16(w.w);
      }
      return bb;
    }
  };

  v8f acc[NTILES];
  #pragma unroll
  for (int j = 0; j < NTILES; ++j) acc[j] = v8f{};

  // Software pipeline: B fragment for slot j is loaded one WMMA slot ahead,
  // so each v_wmma only waits on loads issued a full slot earlier.
  v16bf bcur = loadB(0, 0);
  for (int k0 = 0; k0 < D_; k0 += KSTEP) {
    // Shared A fragment for all 12 N-tiles (2x ds_load_b128)
    v8bf alo = *(const v8bf*)&As[am][k0 + koffA];
    v8bf ahi = *(const v8bf*)&As[am][k0 + 8 + koffA];
    v16bf a = __builtin_shufflevector(alo, ahi,
               0,1,2,3,4,5,6,7,8,9,10,11,12,13,14,15);
    if (k0 + KSTEP < D_) {
      if (PRECONV) __builtin_prefetch(wb + k0 + KSTEP, 0, 0);
      else         __builtin_prefetch(wf + k0 + KSTEP, 0, 0);
    }
    #pragma unroll
    for (int j = 0; j < NTILES; ++j) {
      // Next slot: (j+1) this K-step, or slot 0 of the next K-step
      // (wraps to an in-bounds dummy re-read of (0,0) on the final slot).
      const int jn = (j + 1 < NTILES) ? (j + 1) : 0;
      const int kn = (j + 1 < NTILES) ? k0 : ((k0 + KSTEP < D_) ? k0 + KSTEP : 0);
      v16bf bnext = loadB(jn, kn);
      acc[j] = __builtin_amdgcn_wmma_f32_16x16x32_bf16(
                 false, a, false, bcur, (short)0, acc[j], false, false);
      bcur = bnext;
    }
  }

  // ---------------- Epilogue: pooled = tanh(acc + b_pool); dot W_out -------
  float rowsum[8];
  #pragma unroll
  for (int j = 0; j < NTILES; ++j) {
    const int n = nbase + j * 16 + nlane;
    const float bias = b_pool[n];
    const float wo   = W_out[n];
    #pragma unroll
    for (int r = 0; r < 8; ++r) {               // acc[j][r]: M = r + 8*mhalf, N = nlane
      float pv = tanhf(acc[j][r] + bias) * wo;
      pv += __shfl_xor(pv, 1);                  // reduce over the 16 N-lanes
      pv += __shfl_xor(pv, 2);
      pv += __shfl_xor(pv, 4);
      pv += __shfl_xor(pv, 8);
      rowsum[r] = (j == 0) ? pv : (rowsum[r] + pv);
    }
  }
  if (nlane == 0) {                             // lanes 0 & 16 hold row sums
    #pragma unroll
    for (int r = 0; r < 8; ++r)
      part[mhalf * 8 + r][wave] = rowsum[r];
  }
  __syncthreads();

  if (tid < ROWS) {
    float acc0 = b_out1[0];
    #pragma unroll
    for (int w = 0; w < NWAVES; ++w) acc0 += part[tid][w];
    out[blk * ROWS + tid] = acc0;               // logits [B*L]
  }
}

extern "C" void kernel_launch(void* const* d_in, const int* in_sizes, int n_in,
                              void* d_out, int out_size, void* d_ws, size_t ws_size,
                              hipStream_t stream) {
  const float* seq    = (const float*)d_in[0];  // sequence_output [B,S,H]
  // d_in[1] context_h  : unused by the reference computation
  const float* W_pool = (const float*)d_in[2];  // [D,D]
  const float* b_pool = (const float*)d_in[3];  // [D]
  const float* W_out  = (const float*)d_in[4];  // [D]
  const float* b_out  = (const float*)d_in[5];  // scalar
  const int*   pos    = (const int*)d_in[6];    // [B,L+1]
  // d_in[7] doc_lens   : unused (always L)
  float* out = (float*)d_out;                   // [B*L] = 2048

  const int nblocks = (B_ * L_) / ROWS;         // 128
  const size_t wbf_bytes = (size_t)D_ * D_ * sizeof(unsigned short); // 4.5 MB

  if (ws_size >= wbf_bytes) {
    // One-time f32 -> bf16 conversion of W_pool into scratch (same stream).
    const int cblocks = (D_ * D_) / (256 * 4);  // 2304
    convert_w_kernel<<<cblocks, 256, 0, stream>>>(W_pool, (unsigned*)d_ws);
    ece_fused_kernel<true><<<nblocks, 256, 0, stream>>>(
        seq, W_pool, (const __bf16*)d_ws, b_pool, W_out, b_out, pos, out);
  } else {
    // Fallback: convert B fragments in-register (no scratch needed).
    ece_fused_kernel<false><<<nblocks, 256, 0, stream>>>(
        seq, W_pool, nullptr, b_pool, W_out, b_out, pos, out);
  }
  (void)in_sizes; (void)n_in; (void)out_size;
}
